// ConfTensorProductModel_34591666602691
// MI455X (gfx1250) — matrix operand order repaired and checked by reading, not compiled, not run on hardware
//
#include <hip/hip_runtime.h>
#include <hip/hip_bf16.h>

// ============================================================================
// Equivariant GNN forward for MI455X (gfx1250, wave32, WMMA).
// Per-edge weight-MLP GEMMs (~22 GFLOP) on v_wmma_f32_16x16x32_f16, fused with
// the tensor-product contraction + scatter so the E*1536 weight tensor never
// touches HBM. 32-edge M-tiles: each B fragment feeds two independent WMMA
// accumulator chains (fills the 5-slot f16-WMMA RAW hazard window, halves
// L2 weight traffic per edge).
// ============================================================================

typedef _Float16 half_t;
typedef __attribute__((ext_vector_type(16))) _Float16 v16h;
typedef __attribute__((ext_vector_type(8)))  float    v8f;

// ---- constants -------------------------------------------------------------
#define INV_SQRT3  0.5773502691896258f
#define INV_SQRT6  0.4082482904638631f
#define W121_C1    0.31622776601683794f   // 1/sqrt(10)
#define W121_C2    0.3651483716701107f    // sqrt(2/15)
#define W121_C2H   0.18257418583505536f   // C2/2
#define SQ3        1.7320508075688772f
#define SQ15       3.872983346207417f
#define SQ5H       1.118033988749895f     // sqrt(5)/2
#define SQ15H      1.9364916731037085f    // sqrt(15)/2

// ============================================================================
// WMMA fragment helpers (CDNA5 16x16x32 f16 layouts per ISA 7.12.2)
// ============================================================================
__device__ __forceinline__ v16h load_a_frag(const half_t* A, int stride, int lane, int kstep) {
  const int m  = lane & 15;
  const int hi = lane >> 4;
  const int k0 = kstep * 32 + hi * 8;
  const int k1 = kstep * 32 + 16 + hi * 8;
  const half_t* row = A + m * stride;
  v16h a;
#pragma unroll
  for (int j = 0; j < 8; ++j) { a[j] = row[k0 + j]; a[8 + j] = row[k1 + j]; }
  return a;
}

__device__ __forceinline__ v16h load_b_frag(const half_t* Wt, int kpad, int nbase, int lane, int kstep) {
  const int n  = nbase + (lane & 15);
  const int kh = kstep * 32 + ((lane >> 4) << 3);
  const half_t* p = Wt + n * kpad + kh;
  v16h b;
#pragma unroll
  for (int j = 0; j < 8; ++j) { b[j] = p[j]; b[8 + j] = p[16 + j]; }
  return b;
}

// ============================================================================
// Small elementwise kernels
// ============================================================================
__global__ void zero_kernel(float* p, int n) {
  int i = blockIdx.x * blockDim.x + threadIdx.x;
  if (i < n) p[i] = 0.0f;
}

// Wt[n][k] = (k<K) ? W[k*N+n] : 0   (f32 -> f16, transposed, K padded)
__global__ void convert_wt(const float* __restrict__ W, half_t* __restrict__ Wt,
                           int K, int N, int KPAD) {
  int i = blockIdx.x * blockDim.x + threadIdx.x;
  if (i >= N * KPAD) return;
  int n = i / KPAD, k = i % KPAD;
  Wt[i] = (k < K) ? (half_t)W[k * N + n] : (half_t)0.0f;
}

// temb + node embedding MLP (76->16 relu ->16)
__global__ void node_kernel(const float* __restrict__ x, const float* __restrict__ t,
                            const float* __restrict__ W1, const float* __restrict__ b1,
                            const float* __restrict__ W2, const float* __restrict__ b2,
                            float* __restrict__ temb, float* __restrict__ h0, int N) {
  int n = blockIdx.x * blockDim.x + threadIdx.x;
  if (n >= N) return;
  float tv = t[n];
  float te[32];
#pragma unroll
  for (int i = 0; i < 16; ++i) {
    float f = __expf(-9.210340371976184f * (float)i / 16.0f);
    float a = tv * f;
    te[i] = __sinf(a);
    te[16 + i] = __cosf(a);
  }
#pragma unroll
  for (int i = 0; i < 32; ++i) temb[n * 32 + i] = te[i];
  float attr[76];
#pragma unroll 4
  for (int k = 0; k < 44; ++k) attr[k] = x[n * 44 + k];
#pragma unroll
  for (int k = 0; k < 32; ++k) attr[44 + k] = te[k];
  float hid[16];
  for (int j = 0; j < 16; ++j) {
    float s = b1[j];
#pragma unroll 4
    for (int k = 0; k < 76; ++k) s += attr[k] * W1[k * 16 + j];
    hid[j] = s > 0.0f ? s : 0.0f;
  }
  for (int j = 0; j < 16; ++j) {
    float s = b2[j];
#pragma unroll
    for (int k = 0; k < 16; ++k) s += hid[k] * W2[k * 16 + j];
    h0[n * 16 + j] = s;
  }
}

// per-edge: sph harmonics, gaussian smear, edge MLP (64->16 relu ->16), degree count
__global__ void edge_kernel(const float* __restrict__ pos, const int* __restrict__ ei,
                            const float* __restrict__ temb,
                            const float* __restrict__ W1, const float* __restrict__ b1,
                            const float* __restrict__ W2, const float* __restrict__ b2,
                            float* __restrict__ ea, float* __restrict__ shv,
                            float* __restrict__ cnt, int E) {
  int e = blockIdx.x * blockDim.x + threadIdx.x;
  if (e >= E) return;
  int s = ei[e], d = ei[E + e];
  float vx = pos[d * 3 + 0] - pos[s * 3 + 0];
  float vy = pos[d * 3 + 1] - pos[s * 3 + 1];
  float vz = pos[d * 3 + 2] - pos[s * 3 + 2];
  float r = __fsqrt_rn(vx * vx + vy * vy + vz * vz);
  float inv = 1.0f / fmaxf(r, 1e-9f);
  float X = vx * inv, Y = vy * inv, Z = vz * inv;
  float sh[9];
  sh[0] = 1.0f;
  sh[1] = SQ3 * Y;  sh[2] = SQ3 * Z;  sh[3] = SQ3 * X;
  sh[4] = SQ15 * X * Y;  sh[5] = SQ15 * Y * Z;
  sh[6] = SQ5H * (3.0f * Z * Z - 1.0f);
  sh[7] = SQ15 * X * Z;  sh[8] = SQ15H * (X * X - Y * Y);
#pragma unroll
  for (int i = 0; i < 9; ++i) shv[e * 9 + i] = sh[i];

  float attr[64];
  const float coeff = -0.5338888888888889f;  // -0.5/(30/31)^2
#pragma unroll 4
  for (int i = 0; i < 32; ++i) {
    float mu = 30.0f * (float)i / 31.0f;
    float dd = r - mu;
    attr[i] = __expf(coeff * dd * dd);
  }
#pragma unroll 4
  for (int i = 0; i < 32; ++i) attr[32 + i] = temb[s * 32 + i];
  float hid[16];
  for (int j = 0; j < 16; ++j) {
    float acc = b1[j];
#pragma unroll 4
    for (int k = 0; k < 64; ++k) acc += attr[k] * W1[k * 16 + j];
    hid[j] = acc > 0.0f ? acc : 0.0f;
  }
  for (int j = 0; j < 16; ++j) {
    float acc = b2[j];
#pragma unroll
    for (int k = 0; k < 16; ++k) acc += hid[k] * W2[k * 16 + j];
    ea[e * 16 + j] = acc;
  }
  atomicAdd(&cnt[d], 1.0f);
}

// hout = pad(hin) + acc / max(cnt,1)
__global__ void combine_kernel(const float* __restrict__ hin, int INDIM,
                               const float* __restrict__ acc, const float* __restrict__ cnt,
                               float* __restrict__ hout, int OUTDIM, int N) {
  int i = blockIdx.x * blockDim.x + threadIdx.x;
  if (i >= N * OUTDIM) return;
  int n = i / OUTDIM, d = i % OUTDIM;
  float c = fmaxf(cnt[n], 1.0f);
  float base = (d < INDIM) ? hin[n * INDIM + d] : 0.0f;
  hout[i] = base + acc[i] / c;
}

// out[n,c] = (accF[n,c] + accF[n,c+6]) / max(cnt,1)   (drift,doobs combine)
__global__ void final_kernel(const float* __restrict__ accF, const float* __restrict__ cnt,
                             float* __restrict__ out, int N) {
  int i = blockIdx.x * blockDim.x + threadIdx.x;
  if (i >= N * 6) return;
  int n = i / 6, c = i % 6;
  float cc = fmaxf(cnt[n], 1.0f);
  out[i] = (accF[n * 12 + c] + accF[n * 12 + 6 + c]) / cc;
}

// ============================================================================
// Fused conv kernel, 32-edge tiles (two 16-row WMMA M-tiles sharing B frags):
//   stage -> GEMM1 (48->N1, relu) -> GEMM2 (N1->NW) -> CG contraction (z)
//   -> weight contraction -> atomicAdd scatter into acc[dst]
// Assumes E % 32 == 0 (E = 100000 here -> 3125 blocks).
// ============================================================================
template <int L> struct Cfg;
template <> struct Cfg<0> { enum { HDIM = 16,  N1 = 48, K2PAD = 64, NW = 512,  ZTOT = 64  }; };
template <> struct Cfg<1> { enum { HDIM = 64,  N1 = 48, K2PAD = 64, NW = 1536, ZTOT = 224 }; };
template <> struct Cfg<2> { enum { HDIM = 112, N1 = 32, K2PAD = 32, NW = 224,  ZTOT = 336 }; };

constexpr size_t conv_smem_bytes(int NW, int ZTOT, int HDIM) {
  return (size_t)(32 * ZTOT + 32 * HDIM + 32 * 9) * 4   // zbuf, hrow, shrow (f32)
       + (size_t)(32 * NW + 2 * 32 * 64) * 2            // wbuf (f16) + Xa + A2 (f16)
       + 128;                                           // dstl
}

template <int L>
__global__ __launch_bounds__(256)
void conv_kernel(const float* __restrict__ ea, const float* __restrict__ shv,
                 const int* __restrict__ ei, const float* __restrict__ h,
                 const half_t* __restrict__ W1t, const float* __restrict__ b1,
                 const half_t* __restrict__ W2t, const float* __restrict__ b2,
                 float* __restrict__ acc, int E) {
  constexpr int HDIM = Cfg<L>::HDIM;
  constexpr int N1 = Cfg<L>::N1;
  constexpr int K2PAD = Cfg<L>::K2PAD;
  constexpr int NW = Cfg<L>::NW;
  constexpr int ZTOT = Cfg<L>::ZTOT;
  constexpr int NWT = NW / 16;
  constexpr int N1T = N1 / 16;

  extern __shared__ char smem[];
  float* zbuf   = (float*)smem;              // [32][ZTOT] CG-contracted terms
  float* hrow   = zbuf + 32 * ZTOT;          // [32][HDIM] h[src]
  float* shrow  = hrow + 32 * HDIM;          // [32][9]
  half_t* wbuf  = (half_t*)(shrow + 32 * 9); // [32][NW] generated TP weights (f16)
  half_t* Xa    = wbuf + 32 * NW;            // [32][64] GEMM1 A (f16, K padded)
  half_t* A2    = Xa + 32 * 64;              // [32][64] GEMM2 A (f16, K padded)
  int* dstl     = (int*)(A2 + 32 * 64);      // [32]

  const int tid = threadIdx.x;
  const int lane = tid & 31;
  const int wave = tid >> 5;
  const int ge0 = blockIdx.x * 32;

  // ---- stage 32 edges -------------------------------------------------------
  for (int idx = tid; idx < 32 * 16; idx += 256) {
    int e = idx >> 4, j = idx & 15;
    int ge = ge0 + e;
    int src = ei[ge], dst = ei[E + ge];
    if (j == 0) dstl[e] = dst;
    Xa[e * 64 + j]      = (half_t)ea[ge * 16 + j];
    Xa[e * 64 + 16 + j] = (half_t)h[src * HDIM + j];
    Xa[e * 64 + 32 + j] = (half_t)h[dst * HDIM + j];
    Xa[e * 64 + 48 + j] = (half_t)0.0f;
#pragma unroll
    for (int c = j; c < 64; c += 16) A2[e * 64 + c] = (half_t)0.0f;
    for (int c = j; c < HDIM; c += 16) hrow[e * HDIM + c] = h[src * HDIM + c];
    if (j < 9) shrow[e * 9 + j] = shv[ge * 9 + j];
  }
  __syncthreads();

  // ---- GEMM1: (32x48 pad64) @ W1t -> bias -> relu -> A2 (f16) ---------------
  // wave -> (M-tile, N-tile); 2*N1T wave-tasks (6 for conv, 4 for final)
  if (wave < 2 * N1T) {
    int mt = wave & 1, nt1 = wave >> 1;
    const half_t* Am = Xa + mt * 16 * 64;
    v8f c = {};
#pragma unroll
    for (int ks = 0; ks < 2; ++ks) {
      v16h a = load_a_frag(Am, 64, lane, ks);
      v16h b = load_b_frag(W1t, 64, nt1 * 16, lane, ks);
      c = __builtin_amdgcn_wmma_f32_16x16x32_f16(false, a, false, b, (short)0, c, false, false);
    }
    int col = nt1 * 16 + (lane & 15);
    float bias = b1[col];
    int mb = mt * 16 + (lane >> 4) * 8;
#pragma unroll
    for (int r = 0; r < 8; ++r) {
      float v = c[r] + bias;
      A2[(mb + r) * 64 + col] = (half_t)(v > 0.0f ? v : 0.0f);
    }
  }
  __syncthreads();

  // ---- GEMM2: A2 @ W2t + b2 -> wbuf (f16) -----------------------------------
  // Each wave: one N-tile, both M-tiles -> two independent WMMA chains per B.
  {
    v16h a00 = load_a_frag(A2, 64, lane, 0);
    v16h a10 = load_a_frag(A2 + 16 * 64, 64, lane, 0);
    v16h a01 = a00, a11 = a10;
    if constexpr (K2PAD > 32) {
      a01 = load_a_frag(A2, 64, lane, 1);
      a11 = load_a_frag(A2 + 16 * 64, 64, lane, 1);
    }
    for (int nt = wave; nt < NWT; nt += 8) {
      v8f c0 = {}, c1 = {};
      v16h bf0 = load_b_frag(W2t, K2PAD, nt * 16, lane, 0);
      c0 = __builtin_amdgcn_wmma_f32_16x16x32_f16(false, a00, false, bf0, (short)0, c0, false, false);
      c1 = __builtin_amdgcn_wmma_f32_16x16x32_f16(false, a10, false, bf0, (short)0, c1, false, false);
      if constexpr (K2PAD > 32) {
        v16h bf1 = load_b_frag(W2t, K2PAD, nt * 16, lane, 1);
        c0 = __builtin_amdgcn_wmma_f32_16x16x32_f16(false, a01, false, bf1, (short)0, c0, false, false);
        c1 = __builtin_amdgcn_wmma_f32_16x16x32_f16(false, a11, false, bf1, (short)0, c1, false, false);
      }
      int col = nt * 16 + (lane & 15);
      float bias = b2[col];
      int mb = (lane >> 4) * 8;
#pragma unroll
      for (int r = 0; r < 8; ++r) {
        wbuf[(mb + r) * NW + col]      = (half_t)(c0[r] + bias);
        wbuf[(16 + mb + r) * NW + col] = (half_t)(c1[r] + bias);
      }
    }
  }
  __syncthreads();

  // ---- TP phase A: z[u,*] = sum_ij x[u,i] sh[j] CG[i,j,k] -------------------
  for (int idx = tid; idx < 32 * 16; idx += 256) {
    int e = idx >> 4, u = idx & 15;
    const float* hr = hrow + e * HDIM;
    const float* s = shrow + e * 9;
    float* z = zbuf + e * ZTOT;
    const float s0 = s[0];
    const float s1a = s[1], s1b = s[2], s1c = s[3];
    // l1=1 x l2=2 -> l3=1 mixing matrix (real basis, (y,z,x) order)
    float m00 = -W121_C2H * s[6] - W121_C1 * s[8];
    float m01 = W121_C1 * s[5];
    float m02 = W121_C1 * s[4];
    float m11 = W121_C2 * s[6];
    float m12 = W121_C1 * s[7];
    float m22 = -W121_C2H * s[6] + W121_C1 * s[8];

    if constexpr (L == 0) {
      float x0 = hr[u];
      z[u] = x0 * s0;                                   // (0,0,0)
      z[16 + u * 3 + 0] = x0 * s1a * INV_SQRT3;         // (0,1,1)
      z[16 + u * 3 + 1] = x0 * s1b * INV_SQRT3;
      z[16 + u * 3 + 2] = x0 * s1c * INV_SQRT3;
    } else if constexpr (L == 1) {
      float x0 = hr[u];
      float a0 = hr[16 + u * 3 + 0], a1v = hr[16 + u * 3 + 1], a2 = hr[16 + u * 3 + 2];
      z[u] = x0 * s0;                                   // (0,0,0)
      z[16 + u * 3 + 0] = x0 * s1a * INV_SQRT3;         // (0,1,1)
      z[16 + u * 3 + 1] = x0 * s1b * INV_SQRT3;
      z[16 + u * 3 + 2] = x0 * s1c * INV_SQRT3;
      z[64 + u * 3 + 0] = a0 * s0 * INV_SQRT3;          // (1,0,1)
      z[64 + u * 3 + 1] = a1v * s0 * INV_SQRT3;
      z[64 + u * 3 + 2] = a2 * s0 * INV_SQRT3;
      z[112 + u] = (a0 * s1a + a1v * s1b + a2 * s1c) * INV_SQRT3;   // (1,1,0)
      z[128 + u * 3 + 0] = (a1v * s1c - a2 * s1b) * INV_SQRT6;      // (1,1,1) cross
      z[128 + u * 3 + 1] = (a2 * s1a - a0 * s1c) * INV_SQRT6;
      z[128 + u * 3 + 2] = (a0 * s1b - a1v * s1a) * INV_SQRT6;
      z[176 + u * 3 + 0] = a0 * m00 + a1v * m01 + a2 * m02;         // (1,2,1)
      z[176 + u * 3 + 1] = a0 * m01 + a1v * m11 + a2 * m12;
      z[176 + u * 3 + 2] = a0 * m02 + a1v * m12 + a2 * m22;
    } else {
      float x0 = hr[u];
      float a0 = hr[16 + u * 3 + 0], a1v = hr[16 + u * 3 + 1], a2 = hr[16 + u * 3 + 2];
      float bb0 = hr[64 + u * 3 + 0], bb1 = hr[64 + u * 3 + 1], bb2 = hr[64 + u * 3 + 2];
      z[u * 3 + 0] = x0 * s1a * INV_SQRT3;              // (0,1,0)
      z[u * 3 + 1] = x0 * s1b * INV_SQRT3;
      z[u * 3 + 2] = x0 * s1c * INV_SQRT3;
      z[48 + u * 3 + 0] = a0 * s0 * INV_SQRT3;          // (1,0,0)
      z[48 + u * 3 + 1] = a1v * s0 * INV_SQRT3;
      z[48 + u * 3 + 2] = a2 * s0 * INV_SQRT3;
      z[96 + u * 3 + 0] = (a1v * s1c - a2 * s1b) * INV_SQRT6;  // (1,1,1)
      z[96 + u * 3 + 1] = (a2 * s1a - a0 * s1c) * INV_SQRT6;
      z[96 + u * 3 + 2] = (a0 * s1b - a1v * s1a) * INV_SQRT6;
      z[144 + u * 3 + 0] = a0 * m00 + a1v * m01 + a2 * m02;    // (1,2,0)
      z[144 + u * 3 + 1] = a0 * m01 + a1v * m11 + a2 * m12;
      z[144 + u * 3 + 2] = a0 * m02 + a1v * m12 + a2 * m22;
      z[192 + u * 3 + 0] = bb0 * s0 * INV_SQRT3;        // (2,0,1)
      z[192 + u * 3 + 1] = bb1 * s0 * INV_SQRT3;
      z[192 + u * 3 + 2] = bb2 * s0 * INV_SQRT3;
      z[240 + u * 3 + 0] = (bb1 * s1c - bb2 * s1b) * INV_SQRT6;  // (2,1,0)
      z[240 + u * 3 + 1] = (bb2 * s1a - bb0 * s1c) * INV_SQRT6;
      z[240 + u * 3 + 2] = (bb0 * s1b - bb1 * s1a) * INV_SQRT6;
      z[288 + u * 3 + 0] = bb0 * m00 + bb1 * m01 + bb2 * m02;    // (2,2,1)
      z[288 + u * 3 + 1] = bb0 * m01 + bb1 * m11 + bb2 * m12;
      z[288 + u * 3 + 2] = bb0 * m02 + bb1 * m12 + bb2 * m22;
    }
  }
  __syncthreads();

  // ---- TP phase B: contract with generated weights, scatter to acc[dst] -----
  for (int idx = tid; idx < 32 * 16; idx += 256) {
    int e = idx >> 4, w = idx & 15;
    const half_t* wb = wbuf + e * NW;
    const float* z = zbuf + e * ZTOT;
    int dst = dstl[e];

    if constexpr (L == 0) {
      float o0 = 0.0f, o1x = 0.0f, o1y = 0.0f, o1z = 0.0f;
#pragma unroll 4
      for (int u = 0; u < 16; ++u) {
        float wa = (float)wb[u * 16 + w];
        o0 += wa * z[u];
        float wbv = (float)wb[256 + u * 16 + w];
        o1x += wbv * z[16 + u * 3 + 0];
        o1y += wbv * z[16 + u * 3 + 1];
        o1z += wbv * z[16 + u * 3 + 2];
      }
      atomicAdd(&acc[dst * 64 + w], 0.25f * o0);
      atomicAdd(&acc[dst * 64 + 16 + w * 3 + 0], 0.4330127019f * o1x);
      atomicAdd(&acc[dst * 64 + 16 + w * 3 + 1], 0.4330127019f * o1y);
      atomicAdd(&acc[dst * 64 + 16 + w * 3 + 2], 0.4330127019f * o1z);
    } else if constexpr (L == 1) {
      float o0 = 0.0f, o1[3] = {0, 0, 0}, o2[3] = {0, 0, 0};
#pragma unroll 4
      for (int u = 0; u < 16; ++u) {
        o0 += (float)wb[u * 16 + w] * z[u] + (float)wb[768 + u * 16 + w] * z[112 + u];
        float wa = (float)wb[256 + u * 16 + w];
        float wbp = (float)wb[512 + u * 16 + w];
        float wc = (float)wb[1280 + u * 16 + w];
        float wd = (float)wb[1024 + u * 16 + w];
#pragma unroll
        for (int k = 0; k < 3; ++k) {
          o1[k] += wa * z[16 + u * 3 + k] + wbp * z[64 + u * 3 + k] + wc * z[176 + u * 3 + k];
          o2[k] += wd * z[128 + u * 3 + k];
        }
      }
      atomicAdd(&acc[dst * 112 + w], 0.1767766953f * o0);
#pragma unroll
      for (int k = 0; k < 3; ++k) {
        atomicAdd(&acc[dst * 112 + 16 + w * 3 + k], 0.25f * o1[k]);
        atomicAdd(&acc[dst * 112 + 64 + w * 3 + k], 0.4330127019f * o2[k]);
      }
    } else {
      if (w < 12) {
        int o = w / 6, rem = w % 6, wc = rem / 3, kk = rem % 3;
        float a = 0.0f;
        if (o == 0) {
#pragma unroll 4
          for (int u = 0; u < 16; ++u)
            a += (float)wb[u * 2 + wc] * z[u * 3 + kk]
               + (float)wb[32 + u * 2 + wc] * z[48 + u * 3 + kk]
               + (float)wb[96 + u * 2 + wc] * z[144 + u * 3 + kk]
               + (float)wb[160 + u * 2 + wc] * z[240 + u * 3 + kk];
          a *= 0.2165063509f;
        } else {
#pragma unroll 4
          for (int u = 0; u < 16; ++u)
            a += (float)wb[64 + u * 2 + wc] * z[96 + u * 3 + kk]
               + (float)wb[128 + u * 2 + wc] * z[192 + u * 3 + kk]
               + (float)wb[192 + u * 2 + wc] * z[288 + u * 3 + kk];
          a *= 0.25f;
        }
        atomicAdd(&acc[dst * 12 + w], a);
      }
    }
  }
}

// ============================================================================
// Host launcher
// ============================================================================
extern "C" void kernel_launch(void* const* d_in, const int* in_sizes, int n_in,
                              void* d_out, int out_size, void* d_ws, size_t ws_size,
                              hipStream_t stream) {
  const float* x   = (const float*)d_in[0];
  const float* pos = (const float*)d_in[1];
  const float* t   = (const float*)d_in[2];
  const int*   ei  = (const int*)d_in[3];
  // params in setup_inputs() insertion order: node_emb, edge_emb, conv0_fc, conv1_fc, final_fc
  const float* neW1 = (const float*)d_in[4];
  const float* neb1 = (const float*)d_in[5];
  const float* neW2 = (const float*)d_in[6];
  const float* neb2 = (const float*)d_in[7];
  const float* eeW1 = (const float*)d_in[8];
  const float* eeb1 = (const float*)d_in[9];
  const float* eeW2 = (const float*)d_in[10];
  const float* eeb2 = (const float*)d_in[11];
  const float* c0W1 = (const float*)d_in[12];
  const float* c0b1 = (const float*)d_in[13];
  const float* c0W2 = (const float*)d_in[14];
  const float* c0b2 = (const float*)d_in[15];
  const float* c1W1 = (const float*)d_in[16];
  const float* c1b1 = (const float*)d_in[17];
  const float* c1W2 = (const float*)d_in[18];
  const float* c1b2 = (const float*)d_in[19];
  const float* cfW1 = (const float*)d_in[20];
  const float* cfb1 = (const float*)d_in[21];
  const float* cfW2 = (const float*)d_in[22];
  const float* cfb2 = (const float*)d_in[23];

  const int N = in_sizes[2];
  const int E = in_sizes[3] / 2;

  // ---- workspace layout -----------------------------------------------------
  float* ws   = (float*)d_ws;
  float* temb = ws;                          // N*32
  float* h0   = temb + (size_t)N * 32;       // N*16
  float* h1   = h0 + (size_t)N * 16;         // N*64
  float* h2   = h1 + (size_t)N * 64;         // N*112
  float* eab  = h2 + (size_t)N * 112;        // E*16
  float* shb  = eab + (size_t)E * 16;        // E*9
  float* acc1 = shb + (size_t)E * 9;         // N*64   } contiguous, zeroed together
  float* acc2 = acc1 + (size_t)N * 64;       // N*112  }
  float* accF = acc2 + (size_t)N * 112;      // N*12   }
  float* cnt  = accF + (size_t)N * 12;       // N      }
  half_t* c0w1t = (half_t*)(cnt + N);        // 48*64
  half_t* c0w2t = c0w1t + 48 * 64;           // 512*64
  half_t* c1w1t = c0w2t + 512 * 64;          // 48*64
  half_t* c1w2t = c1w1t + 48 * 64;           // 1536*64
  half_t* cfw1t = c1w2t + 1536 * 64;         // 32*64
  half_t* cfw2t = cfw1t + 32 * 64;           // 224*32

  // ---- weight conversion (f32 -> transposed padded f16) ---------------------
  auto cvt = [&](const float* W, half_t* Wt, int K, int Nc, int KPAD) {
    int n = Nc * KPAD;
    convert_wt<<<(n + 255) / 256, 256, 0, stream>>>(W, Wt, K, Nc, KPAD);
  };
  cvt(c0W1, c0w1t, 48, 48, 64);
  cvt(c0W2, c0w2t, 48, 512, 64);
  cvt(c1W1, c1w1t, 48, 48, 64);
  cvt(c1W2, c1w2t, 48, 1536, 64);
  cvt(cfW1, cfw1t, 48, 32, 64);
  cvt(cfW2, cfw2t, 32, 224, 32);

  // ---- zero accumulators + degree counts ------------------------------------
  {
    int zn = N * (64 + 112 + 12 + 1);
    zero_kernel<<<(zn + 255) / 256, 256, 0, stream>>>(acc1, zn);
  }

  // ---- embeddings -----------------------------------------------------------
  node_kernel<<<(N + 255) / 256, 256, 0, stream>>>(x, t, neW1, neb1, neW2, neb2, temb, h0, N);
  edge_kernel<<<(E + 255) / 256, 256, 0, stream>>>(pos, ei, temb, eeW1, eeb1, eeW2, eeb2,
                                                   eab, shb, cnt, E);

  // ---- fused conv layers (E assumed multiple of 32; E=100000 here) ----------
  const int nblk = E / 32;
  constexpr size_t S0 = conv_smem_bytes(512, 64, 16);
  constexpr size_t S1 = conv_smem_bytes(1536, 224, 64);
  constexpr size_t SF = conv_smem_bytes(224, 336, 112);

  conv_kernel<0><<<nblk, 256, S0, stream>>>(eab, shb, ei, h0, c0w1t, c0b1, c0w2t, c0b2, acc1, E);
  combine_kernel<<<(N * 64 + 255) / 256, 256, 0, stream>>>(h0, 16, acc1, cnt, h1, 64, N);

  conv_kernel<1><<<nblk, 256, S1, stream>>>(eab, shb, ei, h1, c1w1t, c1b1, c1w2t, c1b2, acc2, E);
  combine_kernel<<<(N * 112 + 255) / 256, 256, 0, stream>>>(h1, 64, acc2, cnt, h2, 112, N);

  conv_kernel<2><<<nblk, 256, SF, stream>>>(eab, shb, ei, h2, cfw1t, cfb1, cfw2t, cfb2, accF, E);
  final_kernel<<<(N * 6 + 255) / 256, 256, 0, stream>>>(accF, cnt, (float*)d_out, N);
}
